// VariableLagAttention_90658169684733
// MI455X (gfx1250) — matrix operand order repaired and checked by reading, not compiled, not run on hardware
//
#include <hip/hip_runtime.h>
#include <math.h>

// ---------------------------------------------------------------------------
// VariableLagAttention on MI455X (gfx1250, wave32, WMMA).
// B=8, T=64, N=128, FIN=64, D=64, LAG=8.
//
// All GEMMs via v_wmma_f32_16x16x32_bf16 (f32 accumulate). Flash-style online
// softmax across the 8 lag chunks (padded t-l<0 chunks contribute 128 exp(-m)
// denominator terms analytically, matching the reference's softmax over
// zero-padded scores). K/V chunks are double-buffered in LDS; K is staged with
// GLOBAL_LOAD_ASYNC_TO_LDS_B128 (ASYNCcnt) when the toolchain exposes the
// builtin, with global_prefetch_b8 pulling the chunk after next into WGP$.
// ---------------------------------------------------------------------------

typedef __attribute__((ext_vector_type(16))) __bf16       v16bf;
typedef __attribute__((ext_vector_type(8)))  float        v8f;
typedef __attribute__((ext_vector_type(4)))  unsigned int v4u;
typedef __attribute__((ext_vector_type(4)))  int          v4i;

// Async-to-LDS builtin takes AS(1)/AS(3) pointers to int4 (per hipcc's
// diagnostic: "int __attribute__((vector_size(16))) __device__ *").
typedef __attribute__((address_space(1))) v4i gv4i;   // global int4
typedef __attribute__((address_space(3))) v4i lv4i;   // LDS int4

#if __has_builtin(__builtin_amdgcn_global_load_async_to_lds_b128) && \
    __has_builtin(__builtin_amdgcn_s_wait_asynccnt)
#define USE_ASYNC 1
#else
#define USE_ASYNC 0
#endif

#define B_   8
#define T_   64
#define N_   128
#define FIN_ 64
#define D_   64
#define LAG_ 8

// float -> bf16 (round to nearest even; inputs are well-behaved)
__device__ __forceinline__ unsigned short f2bf(float f) {
  union { float f; unsigned int u; } x; x.f = f;
  unsigned int r = x.u + 0x7FFFu + ((x.u >> 16) & 1u);
  return (unsigned short)(r >> 16);
}

// A-fragment (16x32 bf16) from row-major LDS tile [rows][ld].
// ISA 7.12.2: lane<16: M=lane, K={0..7,16..23}; lane>=16: M=lane-16,
// K={8..15,24..31}.  Two contiguous 16B chunks per lane.
__device__ __forceinline__ v16bf load_frag_a(const unsigned short* base, int ld,
                                             int row0, int col0) {
  const int lane = threadIdx.x & 31;
  const int r  = row0 + (lane & 15);
  const int cb = col0 + ((lane >> 4) << 3);
  union { v4u u[2]; v16bf v; } x;
  x.u[0] = *reinterpret_cast<const v4u*>(base + (size_t)r * ld + cb);
  x.u[1] = *reinterpret_cast<const v4u*>(base + (size_t)r * ld + cb + 16);
  return x.v;
}

// B-fragment (32x16 bf16) from N-major LDS tile [n][k] (k contiguous):
// lane N = lane&15; lane<16 -> K={0..7,16..23}, lane>=16 -> K={8..15,24..31}.
__device__ __forceinline__ v16bf load_frag_b(const unsigned short* base, int ld,
                                             int n0, int k0) {
  const int lane = threadIdx.x & 31;
  const int r  = n0 + (lane & 15);
  const int kb = k0 + ((lane >> 4) << 3);
  union { v4u u[2]; v16bf v; } x;
  x.u[0] = *reinterpret_cast<const v4u*>(base + (size_t)r * ld + kb);
  x.u[1] = *reinterpret_cast<const v4u*>(base + (size_t)r * ld + kb + 16);
  return x.v;
}

__device__ __forceinline__ v8f wmma_bf16(v16bf a, v16bf b, v8f c) {
  // (neg_a, A, neg_b, B, c_mod, C, reuse_a, reuse_b)
  return __builtin_amdgcn_wmma_f32_16x16x32_bf16(false, a, false, b,
                                                 (short)0, c, false, false);
}

// Reductions across the 16-lane half of a wave32 (xor masks < 16 never mix
// halves; matches C-tile row layout: row = i + 8*(lane>=16)).
__device__ __forceinline__ float halfmax(float x) {
#pragma unroll
  for (int m = 1; m < 16; m <<= 1) x = fmaxf(x, __shfl_xor(x, m, 32));
  return x;
}
__device__ __forceinline__ float halfsum(float x) {
#pragma unroll
  for (int m = 1; m < 16; m <<= 1) x += __shfl_xor(x, m, 32);
  return x;
}

// Stage a 128x64 bf16 tile, row-major global -> row-major LDS [128][72].
// 1024 x 16B chunks; async LDS-direct when available.
__device__ __forceinline__ void stage_rm(const unsigned short* __restrict__ g,
                                         unsigned short (*lds)[72], int tid) {
  for (int j = tid; j < 1024; j += 256) {
    const int r = j >> 3, c8 = (j & 7) << 3;
#if USE_ASYNC
    __builtin_amdgcn_global_load_async_to_lds_b128(
        (gv4i*)(g + (size_t)r * 64 + c8), (lv4i*)(&lds[r][c8]), 0, 0);
#else
    *reinterpret_cast<v4u*>(&lds[r][c8]) =
        *reinterpret_cast<const v4u*>(g + (size_t)r * 64 + c8);
#endif
  }
}

// Stage a 128x64 bf16 tile transposed: global [key][d] -> LDS [d][key+pad].
__device__ __forceinline__ void stage_tr(const unsigned short* __restrict__ g,
                                         unsigned short (*lds)[136], int tid) {
  for (int j = tid; j < 1024; j += 256) {
    const int r = j >> 3, c8 = (j & 7) << 3;
    union { v4u v; unsigned short s[8]; } x;
    x.v = *reinterpret_cast<const v4u*>(g + (size_t)r * 64 + c8);
#pragma unroll
    for (int e = 0; e < 8; ++e) lds[c8 + e][r] = x.s[e];
  }
}

// ---------------------------------------------------------------------------
// Kernel 1: Q/K/V = relu(X*W + b), bf16 out. One block = 128 rows.
// ---------------------------------------------------------------------------
__global__ __launch_bounds__(256) void k_qkv(
    const float* __restrict__ X,
    const float* __restrict__ Wq, const float* __restrict__ bq,
    const float* __restrict__ Wk, const float* __restrict__ bk,
    const float* __restrict__ Wv, const float* __restrict__ bv,
    unsigned short* __restrict__ Qb, unsigned short* __restrict__ Kb,
    unsigned short* __restrict__ Vb) {
  __shared__ alignas(16) unsigned short Xs[128][72];     // X tile, row-major
  __shared__ alignas(16) unsigned short Ws[3][64][72];   // W^T (N-major)

  const int tid = threadIdx.x;
  const size_t r0 = (size_t)blockIdx.x * 128;

  // X tile: 2048 float4 chunks -> packed bf16x4 (8B) LDS stores.
  for (int j = tid; j < 2048; j += 256) {
    const int r = j >> 4, c4 = (j & 15) << 2;
    float4 x = *reinterpret_cast<const float4*>(X + (r0 + r) * FIN_ + c4);
    union { unsigned short s[4]; unsigned long long u; } p;
    p.s[0] = f2bf(x.x); p.s[1] = f2bf(x.y); p.s[2] = f2bf(x.z); p.s[3] = f2bf(x.w);
    *reinterpret_cast<unsigned long long*>(&Xs[r][c4]) = p.u;
  }
  // Weights: transpose to N-major while converting.
  for (int j = tid; j < 1024; j += 256) {
    const int k = j >> 4, c4 = (j & 15) << 2;
    float4 q = *reinterpret_cast<const float4*>(Wq + k * D_ + c4);
    float4 kk = *reinterpret_cast<const float4*>(Wk + k * D_ + c4);
    float4 v = *reinterpret_cast<const float4*>(Wv + k * D_ + c4);
    const float qa[4] = {q.x, q.y, q.z, q.w};
    const float ka[4] = {kk.x, kk.y, kk.z, kk.w};
    const float va[4] = {v.x, v.y, v.z, v.w};
#pragma unroll
    for (int e = 0; e < 4; ++e) {
      Ws[0][c4 + e][k] = f2bf(qa[e]);
      Ws[1][c4 + e][k] = f2bf(ka[e]);
      Ws[2][c4 + e][k] = f2bf(va[e]);
    }
  }
  __syncthreads();

  const int wave = tid >> 5, lane = tid & 31;
  const int half = lane >> 4, ln = lane & 15;

  v16bf a0 = load_frag_a(&Xs[0][0], 72, wave * 16, 0);
  v16bf a1 = load_frag_a(&Xs[0][0], 72, wave * 16, 32);

  const float* biases[3] = {bq, bk, bv};
  unsigned short* outs[3] = {Qb, Kb, Vb};

#pragma unroll
  for (int m = 0; m < 3; ++m) {
#pragma unroll
    for (int dt = 0; dt < 4; ++dt) {
      v16bf b0 = load_frag_b(&Ws[m][0][0], 72, dt * 16, 0);
      v16bf b1 = load_frag_b(&Ws[m][0][0], 72, dt * 16, 32);
      v8f c = {};
      c = wmma_bf16(a0, b0, c);
      c = wmma_bf16(a1, b1, c);
      float bias = biases[m][dt * 16 + ln];
#pragma unroll
      for (int i = 0; i < 8; ++i) {
        float v = fmaxf(c[i] + bias, 0.0f);
        size_t row = r0 + wave * 16 + i + half * 8;
        outs[m][row * D_ + dt * 16 + ln] = f2bf(v);
      }
    }
  }
}

// ---------------------------------------------------------------------------
// Kernel 2: lagged attention, one block per (b,t). Double-buffered K/V (async K)
// chunk staging overlapped with WMMA + online softmax.
// ---------------------------------------------------------------------------
__global__ __launch_bounds__(256) void k_attn(
    const unsigned short* __restrict__ Qb, const unsigned short* __restrict__ Kb,
    const unsigned short* __restrict__ Vb, unsigned short* __restrict__ Ab) {
  __shared__ alignas(16) unsigned short Qs[N_][72];        // Q tile
  __shared__ alignas(16) unsigned short Ks[2][N_][72];     // K chunk (dbl buf)
  __shared__ alignas(16) unsigned short Vts[2][D_][136];   // V^T chunk (dbl buf)
  __shared__ alignas(16) unsigned short Ps[8][16][136];    // per-wave P buffer

  const int blk = blockIdx.x;
  const int b = blk / T_, t = blk % T_;
  const int tid = threadIdx.x, wave = tid >> 5, lane = tid & 31;
  const int half = lane >> 4, ln = lane & 15;
  const size_t qbase = ((size_t)(b * T_ + t)) * (size_t)(N_ * D_);

  stage_rm(Qb + qbase, Qs, tid);       // Q tile
  stage_rm(Kb + qbase, Ks[0], tid);    // chunk 0 (lag 0 always valid)
  stage_tr(Vb + qbase, Vts[0], tid);
#if USE_ASYNC
  __builtin_amdgcn_s_wait_asynccnt(0);
#endif
  __syncthreads();

  v16bf aq0 = load_frag_a(&Qs[0][0], 72, wave * 16, 0);
  v16bf aq1 = load_frag_a(&Qs[0][0], 72, wave * 16, 32);

  float m[8], lsum[8];
  v8f o[4] = {};
#pragma unroll
  for (int i = 0; i < 8; ++i) { m[i] = -__builtin_inff(); lsum[i] = 0.0f; }

  for (int l = 0; l < LAG_; ++l) {
    const int cur = l & 1;                    // all control flow block-uniform
    if (l) {
#if USE_ASYNC
      __builtin_amdgcn_s_wait_asynccnt(0);    // chunk l resident in LDS
#endif
      __syncthreads();                        // visible to all waves; buf cur^1
                                              // free (its readers were chunk l-1)
    }
    // Stream chunk l+1 into the other buffer while we compute chunk l.
    if (l + 1 < LAG_ && t - (l + 1) >= 0) {
      const size_t cb = qbase - (size_t)(l + 1) * (N_ * D_);
      stage_rm(Kb + cb, Ks[cur ^ 1], tid);
      stage_tr(Vb + cb, Vts[cur ^ 1], tid);
    }
    // Prefetch chunk l+2 into WGP$ (global_prefetch_b8): 256 x 128B lines
    // cover the 16KB K + 16KB V chunk.
    if (l + 2 < LAG_ && t - (l + 2) >= 0) {
      const size_t cb2 = qbase - (size_t)(l + 2) * (N_ * D_);
      const char* p = (tid < 128)
          ? (const char*)(Kb + cb2) + (size_t)tid * 128
          : (const char*)(Vb + cb2) + (size_t)(tid - 128) * 128;
      __builtin_prefetch(p, 0, 3);
    }

    if (t - l < 0) {                          // zero-padded chunk: scores == 0
#pragma unroll
      for (int i = 0; i < 8; ++i) {
        float mn   = fmaxf(m[i], 0.0f);
        float corr = __expf(m[i] - mn);
        lsum[i] = lsum[i] * corr + 128.0f * __expf(-mn);
        m[i] = mn;
#pragma unroll
        for (int d = 0; d < 4; ++d) o[d][i] *= corr;
      }
      continue;
    }

    // S = (Q * K^T) / 8 : 16x128 strip per wave, 8 column tiles.
    v8f s[8];
#pragma unroll
    for (int nt = 0; nt < 8; ++nt) {
      v16bf b0 = load_frag_b(&Ks[cur][0][0], 72, nt * 16, 0);
      v16bf b1 = load_frag_b(&Ks[cur][0][0], 72, nt * 16, 32);
      v8f c = {};
      c = wmma_bf16(aq0, b0, c);
      c = wmma_bf16(aq1, b1, c);
      s[nt] = c * 0.125f;                     // 1/sqrt(D)
    }

    // Online softmax update (row r = i + 8*half lives in one 16-lane half).
    float mn[8], corr[8];
#pragma unroll
    for (int i = 0; i < 8; ++i) {
      float mx = s[0][i];
#pragma unroll
      for (int nt = 1; nt < 8; ++nt) mx = fmaxf(mx, s[nt][i]);
      mx = halfmax(mx);
      mn[i] = fmaxf(m[i], mx);
    }
#pragma unroll
    for (int i = 0; i < 8; ++i) {
      float rs = 0.0f;
#pragma unroll
      for (int nt = 0; nt < 8; ++nt) {
        float p = __expf(s[nt][i] - mn[i]);
        s[nt][i] = p;
        rs += p;
      }
      float csum = halfsum(rs);
      corr[i] = __expf(m[i] - mn[i]);
      lsum[i] = lsum[i] * corr[i] + csum;
      m[i] = mn[i];
    }

    // C-layout -> A-layout via per-wave LDS buffer (bf16).
#pragma unroll
    for (int i = 0; i < 8; ++i) {
      int rr = i + half * 8;
#pragma unroll
      for (int nt = 0; nt < 8; ++nt)
        Ps[wave][rr][nt * 16 + ln] = f2bf(s[nt][i]);
    }

    // Rescale running output, then O += P * V.
#pragma unroll
    for (int d = 0; d < 4; ++d)
#pragma unroll
      for (int i = 0; i < 8; ++i) o[d][i] *= corr[i];

#pragma unroll
    for (int ks = 0; ks < 4; ++ks) {
      v16bf ap = load_frag_a(&Ps[wave][0][0], 136, 0, ks * 32);
#pragma unroll
      for (int d = 0; d < 4; ++d) {
        v16bf bv = load_frag_b(&Vts[cur][0][0], 136, d * 16, ks * 32);
        o[d] = wmma_bf16(ap, bv, o[d]);
      }
    }
  }

  // Normalize and write agg (bf16).
#pragma unroll
  for (int d = 0; d < 4; ++d)
#pragma unroll
    for (int i = 0; i < 8; ++i) {
      float v = o[d][i] / lsum[i];
      int rr = wave * 16 + i + half * 8;
      Ab[qbase + (size_t)rr * D_ + d * 16 + ln] = f2bf(v);
    }
}

// ---------------------------------------------------------------------------
// Kernel 3: out = relu([Q | agg] * Wo + bo), fp32 output.
// ---------------------------------------------------------------------------
__global__ __launch_bounds__(256) void k_out(
    const unsigned short* __restrict__ Qb, const unsigned short* __restrict__ Ab,
    const float* __restrict__ Wo, const float* __restrict__ bo,
    float* __restrict__ out) {
  __shared__ alignas(16) unsigned short Hs[128][136];    // [rows][2D]
  __shared__ alignas(16) unsigned short Wos[64][136];    // Wo^T (N-major)

  const int tid = threadIdx.x;
  const size_t r0 = (size_t)blockIdx.x * 128;

  for (int j = tid; j < 1024; j += 256) {
    const int r = j >> 3, c8 = (j & 7) << 3;
#if USE_ASYNC
    __builtin_amdgcn_global_load_async_to_lds_b128(
        (gv4i*)(Qb + (r0 + r) * D_ + c8), (lv4i*)(&Hs[r][c8]), 0, 0);
    __builtin_amdgcn_global_load_async_to_lds_b128(
        (gv4i*)(Ab + (r0 + r) * D_ + c8), (lv4i*)(&Hs[r][64 + c8]), 0, 0);
#else
    *reinterpret_cast<v4u*>(&Hs[r][c8]) =
        *reinterpret_cast<const v4u*>(Qb + (r0 + r) * D_ + c8);
    *reinterpret_cast<v4u*>(&Hs[r][64 + c8]) =
        *reinterpret_cast<const v4u*>(Ab + (r0 + r) * D_ + c8);
#endif
  }
  for (int j = tid; j < 2048; j += 256) {                // Wo: [128][64] f32
    const int k = j >> 4, c4 = (j & 15) << 2;
    float4 w = *reinterpret_cast<const float4*>(Wo + k * D_ + c4);
    const float wa[4] = {w.x, w.y, w.z, w.w};
#pragma unroll
    for (int e = 0; e < 4; ++e) Wos[c4 + e][k] = f2bf(wa[e]);
  }
#if USE_ASYNC
  __builtin_amdgcn_s_wait_asynccnt(0);
#endif
  __syncthreads();

  const int wave = tid >> 5, lane = tid & 31;
  const int half = lane >> 4, ln = lane & 15;

  v16bf a[4];
#pragma unroll
  for (int ks = 0; ks < 4; ++ks)
    a[ks] = load_frag_a(&Hs[0][0], 136, wave * 16, ks * 32);

#pragma unroll
  for (int dt = 0; dt < 4; ++dt) {
    v8f c = {};
#pragma unroll
    for (int ks = 0; ks < 4; ++ks) {
      v16bf bb = load_frag_b(&Wos[0][0], 136, dt * 16, ks * 32);
      c = wmma_bf16(a[ks], bb, c);
    }
    float bias = bo[dt * 16 + ln];
#pragma unroll
    for (int i = 0; i < 8; ++i) {
      float v = fmaxf(c[i] + bias, 0.0f);
      out[(r0 + wave * 16 + i + half * 8) * D_ + dt * 16 + ln] = v;
    }
  }
}

// ---------------------------------------------------------------------------
extern "C" void kernel_launch(void* const* d_in, const int* in_sizes, int n_in,
                              void* d_out, int out_size, void* d_ws, size_t ws_size,
                              hipStream_t stream) {
  (void)in_sizes; (void)n_in; (void)out_size; (void)ws_size;
  const float* X  = (const float*)d_in[0];
  const float* Wq = (const float*)d_in[1];
  const float* bq = (const float*)d_in[2];
  const float* Wk = (const float*)d_in[3];
  const float* bk = (const float*)d_in[4];
  const float* Wv = (const float*)d_in[5];
  const float* bv = (const float*)d_in[6];
  const float* Wo = (const float*)d_in[7];
  const float* bo = (const float*)d_in[8];
  float* out = (float*)d_out;

  const size_t NR = (size_t)B_ * T_ * N_;                // 65536 rows
  unsigned short* Qb = (unsigned short*)d_ws;            // bf16 Q
  unsigned short* Kb = Qb + NR * D_;                     // bf16 K
  unsigned short* Vb = Kb + NR * D_;                     // bf16 V
  unsigned short* Ab = Vb + NR * D_;                     // bf16 agg
  // total scratch: 4 * 65536 * 64 * 2 B = 33.6 MB

  k_qkv<<<dim3(NR / 128), dim3(256), 0, stream>>>(X, Wq, bq, Wk, bk, Wv, bv,
                                                  Qb, Kb, Vb);
  k_attn<<<dim3(B_ * T_), dim3(256), 0, stream>>>(Qb, Kb, Vb, Ab);
  k_out<<<dim3(NR / 128), dim3(256), 0, stream>>>(Qb, Ab, Wo, bo, out);
}
